// MSA_Col_Att_41583873360511
// MI455X (gfx1250) — compile-verified, matrix operand order checked by reading, not run on hardware
//
#include <hip/hip_runtime.h>
#include <hip/hip_bf16.h>
#include <math.h>

typedef __attribute__((ext_vector_type(16))) _Float16 v16h;
typedef __attribute__((ext_vector_type(8)))  _Float16 v8h;
typedef __attribute__((ext_vector_type(8)))  float    v8f;

#define S_DIM   256
#define L_DIM   384
#define C_DIM   256
#define H_NUM   8
#define D_DIM   32
#define NORM_F  0.17677669529663688f   // 1/sqrt(32)
#define VT_PITCH 264                    // padded pitch for transposed V (avoids bank conflicts)

// Compiler-level barrier: memory ops cannot be moved across this point.
// Used to keep batched fragment loads ahead of the WMMA chain so the
// waitcnt pass emits staggered partial waits instead of one wait per WMMA.
#define SCHED_BARRIER() asm volatile("" ::: "memory")

// Load a 16-element f16 WMMA fragment from a contiguous 32-element K-run.
// Matches CDNA5 16-bit A/B layout: lane (idx + 16*g) holds K = 8g..8g+7 and 16+8g..23+8g.
__device__ __forceinline__ v16h ld_frag(const _Float16* p, int g) {
    v8h lo = *(const v8h*)(p + 8 * g);
    v8h hi = *(const v8h*)(p + 16 + 8 * g);
    return __builtin_shufflevector(lo, hi, 0,1,2,3,4,5,6,7,8,9,10,11,12,13,14,15);
}

__device__ __forceinline__ v8f wmma_f16(v16h a, v16h b, v8f c) {
    return __builtin_amdgcn_wmma_f32_16x16x32_f16(false, a, false, b, (short)0, c, false, false);
}

// ---------------------------------------------------------------------------
// Kernel 0: pre-swizzle the five 256x256 f32 weight matrices into per-lane
// WMMA B-fragment order: [mat][kstep][ntile][lane][16 halves] (f16).
// ---------------------------------------------------------------------------
__global__ void pack_weights(const float* __restrict__ wq, const float* __restrict__ wk,
                             const float* __restrict__ wv, const float* __restrict__ wg,
                             const float* __restrict__ wo, _Float16* __restrict__ pw) {
    int b    = blockIdx.x;          // b = mat*128 + kstep*16 + ntile
    int lane = threadIdx.x;
    int mat  = b >> 7;
    int rem  = b & 127;
    int kstep = rem >> 4;
    int ntile = rem & 15;
    const float* W = (mat == 0) ? wq : (mat == 1) ? wk : (mat == 2) ? wv : (mat == 3) ? wg : wo;
    int n = lane & 15, g = lane >> 4;
    int col = ntile * 16 + n;
    _Float16* dst = pw + (size_t)b * 512 + lane * 16;
#pragma unroll
    for (int e = 0; e < 16; ++e) {
        int kk = kstep * 32 + ((e < 8) ? (8 * g + e) : (16 + 8 * g + (e - 8)));
        dst[e] = (_Float16)W[kk * 256 + col];
    }
}

// ---------------------------------------------------------------------------
// Kernel 1: LayerNorm + Q/K/V/Gate projections. 64-row blocks so each
// B-fragment fetched from L2 is reused across 4 row sub-tiles (4x less
// weight traffic than 16-row blocks).
//   q,k,v -> [h][l][S][d] f16 panels (q pre-scaled by 1/sqrt(D))
//   gate  -> sigmoid(m@wg + bg), natural [s*L+l][256] f16 layout
// ---------------------------------------------------------------------------
__global__ void __launch_bounds__(256) ln_proj_kernel(
        const float* __restrict__ msa, const float* __restrict__ gamma,
        const float* __restrict__ beta, const float* __restrict__ bg,
        const _Float16* __restrict__ pw,
        _Float16* __restrict__ qws, _Float16* __restrict__ kws,
        _Float16* __restrict__ vws, _Float16* __restrict__ gws) {
    __shared__ __align__(16) _Float16 smx[64 * 256];   // 32 KB normalized tile
    __shared__ float redS[256];
    __shared__ float redQ[256];

    int tid = threadIdx.x;
    int r0  = blockIdx.x * 64;     // 64 | 384 -> all rows share one s index
    int row = tid >> 2, seg = tid & 3;   // 4 threads per row, 64 cols each

    // ---- LayerNorm of 64 rows into f16 LDS tile ----
    const float* src = msa + (size_t)(r0 + row) * 256 + seg * 64;
    float xv[64];
    float s1 = 0.f, s2 = 0.f;
#pragma unroll
    for (int j = 0; j < 64; ++j) { float x = src[j]; xv[j] = x; s1 += x; s2 += x * x; }
    redS[tid] = s1; redQ[tid] = s2;
    __syncthreads();
    float ts = 0.f, tq = 0.f;
#pragma unroll
    for (int j = 0; j < 4; ++j) { ts += redS[row * 4 + j]; tq += redQ[row * 4 + j]; }
    float mu  = ts * (1.0f / 256.0f);
    float var = tq * (1.0f / 256.0f) - mu * mu;
    float inv = rsqrtf(var + 1e-5f);
#pragma unroll
    for (int j = 0; j < 64; ++j) {
        int col = seg * 64 + j;
        smx[row * 256 + col] = (_Float16)((xv[j] - mu) * inv * gamma[col] + beta[col]);
    }
    __syncthreads();

    // ---- GEMM: 64 N-tile jobs over 8 waves; B kept in registers per job,
    //      reused by the 4 row sub-tiles. ----
    int wave = tid >> 5, lane = tid & 31;
    int n = lane & 15, g = lane >> 4;
    const v16h* pwv = (const v16h*)pw;
    int s_idx = r0 / L_DIM;
    int l0    = r0 % L_DIM;

    for (int nt = wave; nt < 64; nt += 8) {
        int mat = nt >> 4, ct = nt & 15;
        // Batch-load all 8 B-fragments (16 b128 loads in flight).
        v16h bfr[8];
#pragma unroll
        for (int ks = 0; ks < 8; ++ks)
            bfr[ks] = pwv[((mat * 8 + ks) * 16 + ct) * 32 + lane];
        SCHED_BARRIER();

        int col = ct * 16 + n;
        for (int sub = 0; sub < 4; ++sub) {
            // Batch-load the 8 A-fragments for this 16-row sub-tile.
            v16h afr[8];
#pragma unroll
            for (int ks = 0; ks < 8; ++ks)
                afr[ks] = ld_frag(&smx[(sub * 16 + n) * 256 + ks * 32], g);
            SCHED_BARRIER();
            v8f acc = {};
#pragma unroll
            for (int ks = 0; ks < 8; ++ks)
                acc = wmma_f16(afr[ks], bfr[ks], acc);

            int lrbase = sub * 16 + 8 * g;
            if (mat == 3) {
                float bias = bg[col];
#pragma unroll
                for (int t = 0; t < 8; ++t) {
                    float val  = acc[t] + bias;
                    float gate = 1.0f / (1.0f + __expf(-val));
                    gws[(size_t)(r0 + lrbase + t) * 256 + col] = (_Float16)gate;
                }
            } else {
                _Float16* dstb = (mat == 0) ? qws : (mat == 1) ? kws : vws;
                float scl = (mat == 0) ? NORM_F : 1.0f;
                int h = col >> 5, d = col & 31;
#pragma unroll
                for (int t = 0; t < 8; ++t) {
                    int l = l0 + lrbase + t;
                    dstb[((size_t)(h * L_DIM + l) * S_DIM + s_idx) * D_DIM + d] =
                        (_Float16)(acc[t] * scl);
                }
            }
        }
    }
}

// ---------------------------------------------------------------------------
// Kernel 2: column attention, one block per (h, c). Flash-style softmax.
// Each of 8 waves owns 32 query rows (two 16-row WMMA tiles, sequential).
// ---------------------------------------------------------------------------
__global__ void __launch_bounds__(256) attn_kernel(
        const _Float16* __restrict__ qws, const _Float16* __restrict__ kws,
        const _Float16* __restrict__ vws, _Float16* __restrict__ ows) {
    __shared__ __align__(16) _Float16 sQ[256 * 32];
    __shared__ __align__(16) _Float16 sK[256 * 32];
    __shared__ __align__(16) _Float16 sVt[32 * VT_PITCH];
    __shared__ __align__(16) _Float16 sP[8 * 16 * 32];   // per-wave P repack buffers

    int tid = threadIdx.x;
    int b   = blockIdx.x;
    int h   = b / L_DIM, c = b % L_DIM;
    size_t base = (size_t)(h * L_DIM + c) * (S_DIM * D_DIM);

    // Stage Q, K row-major; V transposed (sVt[d][s]) for contiguous-K B fragments.
    const uint4* gq = (const uint4*)(qws + base);
    const uint4* gk = (const uint4*)(kws + base);
    uint4* lq = (uint4*)sQ;
    uint4* lk = (uint4*)sK;
    for (int i = tid; i < 1024; i += 256) { lq[i] = gq[i]; lk[i] = gk[i]; }
    const _Float16* gv = vws + base;
    for (int i = tid; i < 8192; i += 256)
        sVt[(i & 31) * VT_PITCH + (i >> 5)] = gv[i];
    __syncthreads();

    int wave = tid >> 5, lane = tid & 31;
    int n = lane & 15, g = lane >> 4;
    _Float16* sPw = sP + wave * 512;

    for (int it = 0; it < 2; ++it) {
        int i0 = wave * 32 + it * 16;
        v16h qa = ld_frag(&sQ[(i0 + n) * 32], g);   // Q tile: A fragment, K = D = 32
        v8f O0 = {}, O1 = {};
        float rowm[8], rowl[8];
#pragma unroll
        for (int t = 0; t < 8; ++t) { rowm[t] = -1e30f; rowl[t] = 0.f; }

        // prime the K-fragment pipeline
        v16h kb0 = ld_frag(&sK[n * 32], g);
        v16h kb1 = ld_frag(&sK[(16 + n) * 32], g);

        for (int j0 = 0; j0 < 256; j0 += 32) {
            // scores: S = Q * K^T  (two 16-col tiles)
            v8f Sc0 = {}, Sc1 = {};
            Sc0 = wmma_f16(qa, kb0, Sc0);
            Sc1 = wmma_f16(qa, kb1, Sc1);

            // issue V-fragment loads + next-step K prefetch so ds latency
            // overlaps the softmax VALU work below
            v16h vb0 = ld_frag(&sVt[n * VT_PITCH + j0], g);
            v16h vb1 = ld_frag(&sVt[(16 + n) * VT_PITCH + j0], g);
            if (j0 + 32 < 256) {
                kb0 = ld_frag(&sK[(j0 + 32 + n) * 32], g);
                kb1 = ld_frag(&sK[(j0 + 48 + n) * 32], g);
            }

            // online softmax update; row (t+8g) lives in the 16-lane group of g
#pragma unroll
            for (int t = 0; t < 8; ++t) {
                float a0 = Sc0[t], a1 = Sc1[t];
                float mx = fmaxf(a0, a1);
                mx = fmaxf(mx, __shfl_xor(mx, 1, 32));
                mx = fmaxf(mx, __shfl_xor(mx, 2, 32));
                mx = fmaxf(mx, __shfl_xor(mx, 4, 32));
                mx = fmaxf(mx, __shfl_xor(mx, 8, 32));
                float mnew = fmaxf(rowm[t], mx);
                float scl  = __expf(rowm[t] - mnew);
                float p0 = __expf(a0 - mnew);
                float p1 = __expf(a1 - mnew);
                float sm2 = p0 + p1;
                sm2 += __shfl_xor(sm2, 1, 32);
                sm2 += __shfl_xor(sm2, 2, 32);
                sm2 += __shfl_xor(sm2, 4, 32);
                sm2 += __shfl_xor(sm2, 8, 32);
                rowl[t] = rowl[t] * scl + sm2;
                rowm[t] = mnew;
                O0[t] *= scl;
                O1[t] *= scl;
                sPw[(t + 8 * g) * 32 + n]      = (_Float16)p0;   // C-layout -> LDS
                sPw[(t + 8 * g) * 32 + 16 + n] = (_Float16)p1;
            }
            asm volatile("s_wait_dscnt 0" ::: "memory");          // P stores visible to loads

            // O += P * V  (P as A fragment, V^T rows give contiguous-K B fragments)
            v16h pa = ld_frag(&sPw[n * 32], g);
            O0 = wmma_f16(pa, vb0, O0);
            O1 = wmma_f16(pa, vb1, O1);
        }

        // finalize: divide by row sums, store to natural [s*L+c][h*32+d] layout
#pragma unroll
        for (int t = 0; t < 8; ++t) {
            float invl = 1.0f / rowl[t];
            int srow = i0 + t + 8 * g;
            size_t o = (size_t)(srow * L_DIM + c) * 256 + h * 32;
            ows[o + n]      = (_Float16)(O0[t] * invl);
            ows[o + 16 + n] = (_Float16)(O1[t] * invl);
        }
    }
}

// ---------------------------------------------------------------------------
// Kernel 3: out = (attn_out * gate) @ wo + bo   (f32 output)
// 64-row blocks; wo B-fragments reused across 4 row sub-tiles.
// ---------------------------------------------------------------------------
__global__ void __launch_bounds__(256) out_proj_kernel(
        const _Float16* __restrict__ ows, const _Float16* __restrict__ gws,
        const _Float16* __restrict__ pw, const float* __restrict__ bo,
        float* __restrict__ out) {
    __shared__ __align__(16) _Float16 sA[64 * 256];    // 32 KB gated-activation tile

    int tid = threadIdx.x;
    int r0  = blockIdx.x * 64;
    int row = tid >> 2, seg = tid & 3;
    size_t rb = (size_t)(r0 + row) * 256 + seg * 64;
    const v8h* po = (const v8h*)(ows + rb);
    const v8h* pg = (const v8h*)(gws + rb);
    v8h* pa = (v8h*)(&sA[row * 256 + seg * 64]);
#pragma unroll
    for (int j = 0; j < 8; ++j)
        pa[j] = po[j] * pg[j];           // elementwise gate in f16
    __syncthreads();

    int wave = tid >> 5, lane = tid & 31;
    int n = lane & 15, g = lane >> 4;
    const v16h* pwv = (const v16h*)pw;

    for (int ct = wave * 2; ct < wave * 2 + 2; ++ct) {
        v16h bfr[8];
#pragma unroll
        for (int ks = 0; ks < 8; ++ks)
            bfr[ks] = pwv[((4 * 8 + ks) * 16 + ct) * 32 + lane];
        SCHED_BARRIER();

        int col = ct * 16 + n;
        float bias = bo[col];
        for (int sub = 0; sub < 4; ++sub) {
            v16h afr[8];
#pragma unroll
            for (int ks = 0; ks < 8; ++ks)
                afr[ks] = ld_frag(&sA[(sub * 16 + n) * 256 + ks * 32], g);
            SCHED_BARRIER();
            v8f acc = {};
#pragma unroll
            for (int ks = 0; ks < 8; ++ks)
                acc = wmma_f16(afr[ks], bfr[ks], acc);

            int lrbase = sub * 16 + 8 * g;
#pragma unroll
            for (int t = 0; t < 8; ++t)
                out[(size_t)(r0 + lrbase + t) * 256 + col] = acc[t] + bias;
        }
    }
}

// ---------------------------------------------------------------------------
extern "C" void kernel_launch(void* const* d_in, const int* in_sizes, int n_in,
                              void* d_out, int out_size, void* d_ws, size_t ws_size,
                              hipStream_t stream) {
    const float* msa   = (const float*)d_in[0];
    const float* gamma = (const float*)d_in[1];
    const float* beta  = (const float*)d_in[2];
    const float* wq    = (const float*)d_in[3];
    const float* wk    = (const float*)d_in[4];
    const float* wv    = (const float*)d_in[5];
    const float* wg    = (const float*)d_in[6];
    const float* bg    = (const float*)d_in[7];
    const float* wo    = (const float*)d_in[8];
    const float* bo    = (const float*)d_in[9];
    float* out = (float*)d_out;

    char* ws = (char*)d_ws;
    const size_t QSZ = (size_t)H_NUM * L_DIM * S_DIM * D_DIM;   // 25,165,824 f16 elements
    _Float16* pw  = (_Float16*)ws;                // 640 KB packed weights
    _Float16* qws = (_Float16*)(ws + (1 << 20));  // 48 MB each
    _Float16* kws = qws + QSZ;
    _Float16* vws = kws + QSZ;
    _Float16* gws = vws + QSZ;
    _Float16* ows = gws + QSZ;

    pack_weights<<<640, 32, 0, stream>>>(wq, wk, wv, wg, wo, pw);
    ln_proj_kernel<<<(S_DIM * L_DIM) / 64, 256, 0, stream>>>(msa, gamma, beta, bg, pw,
                                                             qws, kws, vws, gws);
    attn_kernel<<<H_NUM * L_DIM, 256, 0, stream>>>(qws, kws, vws, ows);
    out_proj_kernel<<<(S_DIM * L_DIM) / 64, 256, 0, stream>>>(ows, gws, pw, bo, out);
}
